// MultiHeadSelfAttention_17111149707931
// MI455X (gfx1250) — matrix operand order repaired and checked by reading, not compiled, not run on hardware
//
#include <hip/hip_runtime.h>
#include <hip/hip_bf16.h>

// ---------------------------------------------------------------------------
// Star-Transformer-style attention for MI455X (gfx1250, wave32, WMMA).
// GEMMs: bf16 V_WMMA_F32_16X16X32_BF16 with f32 accumulation, fed by
// double-buffered GLOBAL_LOAD_ASYNC_TO_LDS_B128 (ASYNCcnt) pipelines.
// ---------------------------------------------------------------------------

typedef __attribute__((ext_vector_type(8)))  float  v8f;
typedef __attribute__((ext_vector_type(16))) __bf16 v16bf;

static constexpr int Bx   = 8;
static constexpr int Dd   = 512;
static constexpr int Ll   = 4096;
static constexpr int LP1  = Ll + 1;      // 4097
static constexpr int NH   = 8;
static constexpr int HD   = 64;
static constexpr float SCALE = 0.125f;   // 1/sqrt(64)

// GEMM tiling
static constexpr int TM  = 128;          // rows (positions) per workgroup
static constexpr int TN  = 64;           // cols (output features) per workgroup
static constexpr int TK  = 32;           // K per WMMA step (bf16)
static constexpr int LDT = 40;           // LDS row stride (bf16): 80B -> 16B-aligned
                                         // rows AND conflict-free frag reads
static constexpr int NKIT = Dd / TK;     // 16 K-steps
static constexpr int NLBLK_QKV = 33;     // ceil(4097/128)
static constexpr int NLBLK_OUT = 32;     // 4096/128

union Frag { unsigned int u[8]; v16bf v; };

__device__ __forceinline__ v8f vzero8() {
  v8f z = {0.f,0.f,0.f,0.f,0.f,0.f,0.f,0.f};
  return z;
}

// ---- CDNA5 async global<->LDS (ASYNCcnt) --------------------------------
__device__ __forceinline__ unsigned lds_off32(const void* p) {
  // generic pointer to __shared__ : low 32 bits are the LDS byte offset
  return (unsigned)(size_t)p;
}
__device__ __forceinline__ void async_load_b128(const void* lds, const void* gaddr) {
  asm volatile("global_load_async_to_lds_b128 %0, %1, off"
               :: "v"(lds_off32(lds)), "v"((unsigned long long)(size_t)gaddr)
               : "memory");
}
__device__ __forceinline__ void async_store_b128(void* gaddr, const void* lds) {
  asm volatile("global_store_async_from_lds_b128 %0, %1, off"
               :: "v"((unsigned long long)(size_t)gaddr), "v"(lds_off32(lds))
               : "memory");
}
__device__ __forceinline__ void wait_async0() {
  asm volatile("s_wait_asynccnt 0x0" ::: "memory");
}

// A fragment: 16x32 bf16, rows m = lane&15, K halves per ISA 7.12.2:
//   VGPR v<4 : K = kh + 2v,   VGPR v>=4 : K = 16 + kh + 2(v-4), kh = (lane>>4)*8
__device__ __forceinline__ v16bf load_afrag(const __bf16* base, int lane) {
  const int m  = lane & 15;
  const int kh = (lane >> 4) << 3;
  const unsigned int* p = (const unsigned int*)(base + (size_t)m * LDT);
  Frag f;
#pragma unroll
  for (int v = 0; v < 8; ++v) {
    const int c = ((v & 4) ? 16 : 0) + kh + ((v & 3) << 1);
    f.u[v] = p[c >> 1];
  }
  return f.v;
}

// B fragment: 32x16 bf16, col n = lane&15, K same as A: v<4 : kh + 2v, v>=4 : 16 + kh + 2(v-4), kh = (lane>>4)*8 (+1 packed).
__device__ __forceinline__ v16bf load_bfrag(const __bf16* base, int lane) {
  const int n  = lane & 15;
  const int kh = (lane >> 4) << 3;
  const unsigned int* p = (const unsigned int*)(base + (size_t)n * LDT);
  Frag f;
#pragma unroll
  for (int v = 0; v < 8; ++v) f.u[v] = p[(((v & 4) ? 16 : 0) + kh + ((v & 3) << 1)) >> 1];
  return f.v;
}

__device__ __forceinline__ v8f wmma_bf16(v16bf a, v16bf b, v8f c) {
  return __builtin_amdgcn_wmma_f32_16x16x32_bf16(false, a, false, b, (short)0, c,
                                                 false, false);
}

// ---------------------------------------------------------------------------
// Kernel 0a: weights -> bf16, transposed to [e][k] for contiguous B tiles.
// ---------------------------------------------------------------------------
__global__ void prep_weights(const float* __restrict__ Wq, const float* __restrict__ Wk,
                             const float* __restrict__ Wv, const float* __restrict__ WO,
                             __bf16* __restrict__ wqT, __bf16* __restrict__ wkT,
                             __bf16* __restrict__ wvT, __bf16* __restrict__ woT) {
  const int idx = blockIdx.x * 256 + threadIdx.x;      // over D*D, coalesced on e
  if (idx >= Dd * Dd) return;
  const int kk = idx >> 9;
  const int e  = idx & (Dd - 1);
  const size_t t = (size_t)e * Dd + kk;
  wqT[t] = (__bf16)Wq[idx];
  wkT[t] = (__bf16)Wk[idx];
  wvT[t] = (__bf16)Wv[idx];
  woT[t] = (__bf16)WO[idx];
}

// ---------------------------------------------------------------------------
// Kernel 0b: xy = concat(x, y) transposed to (B, L+1, D) bf16 via LDS tiles.
// Makes the GEMM A-tiles pure byte copies (async-eligible) and halves traffic.
// ---------------------------------------------------------------------------
__global__ __launch_bounds__(256) void xy_to_bf16(const float* __restrict__ x,
                                                  const float* __restrict__ y,
                                                  __bf16* __restrict__ xyb) {
  __shared__ float t[64][65];                  // pad 65 -> conflict-free transpose
  int blk = blockIdx.x;
  const int dblk = blk & 7;            blk >>= 3;
  const int lblk = blk % 65;           // ceil(4097/64)
  const int b    = blk / 65;
  const int l0 = lblk * 64, d0 = dblk * 64;
  const int tid = threadIdx.x;
#pragma unroll
  for (int i = 0; i < 16; ++i) {
    const int idx  = tid + (i << 8);
    const int loff = idx & 63, doff = idx >> 6;   // coalesced on l
    const int l = l0 + loff;
    float val = 0.f;
    if (l < Ll)       val = x[((size_t)b * Dd + d0 + doff) * Ll + l];
    else if (l == Ll) val = y[b * Dd + d0 + doff];
    t[doff][loff] = val;
  }
  __syncthreads();
#pragma unroll
  for (int i = 0; i < 16; ++i) {
    const int idx  = tid + (i << 8);
    const int doff = idx & 63, loff = idx >> 6;   // coalesced on d
    const int l = l0 + loff;
    if (l < LP1)
      xyb[((size_t)b * LP1 + l) * Dd + d0 + doff] = (__bf16)t[doff][loff];
  }
}

// ---------------------------------------------------------------------------
// Kernel 1: fused QKV GEMM, async double-buffered tiles.
// q,k,v out as (B, L+1, D) f32.
// ---------------------------------------------------------------------------
__global__ __launch_bounds__(256) void qkv_gemm(
    const __bf16* __restrict__ xyb,
    const __bf16* __restrict__ wqT, const __bf16* __restrict__ wkT,
    const __bf16* __restrict__ wvT,
    float* __restrict__ q, float* __restrict__ k, float* __restrict__ v) {
  __shared__ __bf16 sA[2][TM * LDT];
  __shared__ __bf16 sB[2][3][TN * LDT];

  const int tid  = threadIdx.x;
  const int wid  = tid >> 5;
  const int lane = tid & 31;
  int blk = blockIdx.x;
  const int eblk = blk & 7;             blk >>= 3;
  const int lblk = blk % NLBLK_QKV;
  const int b    = blk / NLBLK_QKV;
  const int l0   = lblk * TM;
  const int e0   = eblk * TN;

  const __bf16* wT[3] = {wqT, wkT, wvT};

  v8f accQ[4], accK[4], accV[4];
#pragma unroll
  for (int nt = 0; nt < 4; ++nt) { accQ[nt] = vzero8(); accK[nt] = vzero8(); accV[nt] = vzero8(); }

  auto issue_tiles = [&](int buf, int k0) {
    // A tile: 128 rows x 64B -> 512 x B128 chunks, 2 per thread
#pragma unroll
    for (int i = 0; i < 2; ++i) {
      const int idx = tid + (i << 8);
      const int row = idx >> 2, ch = idx & 3;
      const int l = l0 + row;
      if (l < LP1)
        async_load_b128(&sA[buf][row * LDT + (ch << 3)],
                        xyb + ((size_t)b * LP1 + l) * Dd + k0 + (ch << 3));
    }
    // B tiles: 3 x (64 rows x 64B) -> 256 chunks each, 1 per thread per weight
    const int row = tid >> 2, ch = tid & 3;
#pragma unroll
    for (int w = 0; w < 3; ++w)
      async_load_b128(&sB[buf][w][row * LDT + (ch << 3)],
                      wT[w] + (size_t)(e0 + row) * Dd + k0 + (ch << 3));
  };

  issue_tiles(0, 0);
  for (int it = 0; it < NKIT; ++it) {
    wait_async0();          // this wave's outstanding async tile complete
    __syncthreads();        // whole WG: tile ready, prev compute done
    if (it + 1 < NKIT) issue_tiles((it + 1) & 1, (it + 1) * TK);
    const int buf = it & 1;
    const v16bf af = load_afrag(&sA[buf][(wid << 4) * LDT], lane);
#pragma unroll
    for (int nt = 0; nt < 4; ++nt) {
      const v16bf bq = load_bfrag(&sB[buf][0][(nt << 4) * LDT], lane);
      const v16bf bk = load_bfrag(&sB[buf][1][(nt << 4) * LDT], lane);
      const v16bf bv = load_bfrag(&sB[buf][2][(nt << 4) * LDT], lane);
      accQ[nt] = wmma_bf16(af, bq, accQ[nt]);
      accK[nt] = wmma_bf16(af, bk, accK[nt]);
      accV[nt] = wmma_bf16(af, bv, accV[nt]);
    }
  }

  // C layout: VGPR r -> (M=r, N=lane) lanes0-15, (M=r+8, N=lane-16)
  const int m0  = wid << 4;
  const int nlo = lane & 15;
  const int mhi = (lane >> 4) << 3;
#pragma unroll
  for (int nt = 0; nt < 4; ++nt) {
#pragma unroll
    for (int r = 0; r < 8; ++r) {
      const int l = l0 + m0 + r + mhi;
      if (l <= Ll) {
        const size_t o = ((size_t)b * LP1 + l) * Dd + e0 + (nt << 4) + nlo;
        q[o] = accQ[nt][r];
        k[o] = accK[nt][r];
        v[o] = accV[nt][r];
      }
    }
  }
}

// ---------------------------------------------------------------------------
// Kernel 2: windowed node attention (3-window + relay), exact softmax over 4.
// One thread per (b, head, l).  att output: (B, L, D) bf16 (feeds next WMMA).
// ---------------------------------------------------------------------------
__global__ __launch_bounds__(256) void node_attn(
    const float* __restrict__ q, const float* __restrict__ k,
    const float* __restrict__ v, __bf16* __restrict__ att) {
  const int gid = blockIdx.x * 256 + threadIdx.x;
  const int l = gid & (Ll - 1);
  const int n = (gid >> 12) & (NH - 1);
  const int b = gid >> 15;

  const size_t base = (size_t)b * LP1 * Dd + n * HD;
  const float4* q4  = (const float4*)(q + base + (size_t)l * Dd);
  const float4* km1 = (l > 0)      ? (const float4*)(k + base + (size_t)(l - 1) * Dd) : nullptr;
  const float4* k00 = (const float4*)(k + base + (size_t)l * Dd);
  const float4* kp1 = (l < Ll - 1) ? (const float4*)(k + base + (size_t)(l + 1) * Dd) : nullptr;
  const float4* krr = (const float4*)(k + base + (size_t)Ll * Dd);

  float4 qq[16];
#pragma unroll
  for (int j = 0; j < 16; ++j) qq[j] = q4[j];

  float s0 = 0.f, s1 = 0.f, s2 = 0.f, s3 = 0.f;
#pragma unroll
  for (int j = 0; j < 16; ++j) {
    const float4 a = qq[j];
    if (km1) { const float4 t = km1[j]; s0 += a.x*t.x + a.y*t.y + a.z*t.z + a.w*t.w; }
    {         const float4 t = k00[j]; s1 += a.x*t.x + a.y*t.y + a.z*t.z + a.w*t.w; }
    if (kp1) { const float4 t = kp1[j]; s2 += a.x*t.x + a.y*t.y + a.z*t.z + a.w*t.w; }
    {         const float4 t = krr[j]; s3 += a.x*t.x + a.y*t.y + a.z*t.z + a.w*t.w; }
  }
  s0 *= SCALE; s1 *= SCALE; s2 *= SCALE; s3 *= SCALE;   // padded windows keep score 0
  const float m  = fmaxf(fmaxf(s0, s1), fmaxf(s2, s3));
  float p0 = __expf(s0 - m), p1 = __expf(s1 - m), p2 = __expf(s2 - m), p3 = __expf(s3 - m);
  const float inv = 1.f / (p0 + p1 + p2 + p3);
  p0 *= inv; p1 *= inv; p2 *= inv; p3 *= inv;

  const float4* vm1 = (l > 0)      ? (const float4*)(v + base + (size_t)(l - 1) * Dd) : nullptr;
  const float4* v00 = (const float4*)(v + base + (size_t)l * Dd);
  const float4* vp1 = (l < Ll - 1) ? (const float4*)(v + base + (size_t)(l + 1) * Dd) : nullptr;
  const float4* vrr = (const float4*)(v + base + (size_t)Ll * Dd);
  __bf16* ob = att + ((size_t)b * Ll + l) * Dd + n * HD;
#pragma unroll
  for (int j = 0; j < 16; ++j) {
    float4 o = {0.f, 0.f, 0.f, 0.f};
    if (vm1) { const float4 t = vm1[j]; o.x += p0*t.x; o.y += p0*t.y; o.z += p0*t.z; o.w += p0*t.w; }
    {         const float4 t = v00[j]; o.x += p1*t.x; o.y += p1*t.y; o.z += p1*t.z; o.w += p1*t.w; }
    if (vp1) { const float4 t = vp1[j]; o.x += p2*t.x; o.y += p2*t.y; o.z += p2*t.z; o.w += p2*t.w; }
    {         const float4 t = vrr[j]; o.x += p3*t.x; o.y += p3*t.y; o.z += p3*t.z; o.w += p3*t.w; }
    union { __bf16 h[4]; unsigned long long u; } pk;
    pk.h[0] = (__bf16)o.x; pk.h[1] = (__bf16)o.y;
    pk.h[2] = (__bf16)o.z; pk.h[3] = (__bf16)o.w;
    *reinterpret_cast<unsigned long long*>(ob + (j << 2)) = pk.u;
  }
}

// ---------------------------------------------------------------------------
// Kernel 3: relay attention — one WG per (b, head), softmax over 4097 keys.
// ---------------------------------------------------------------------------
__global__ __launch_bounds__(256) void relay_attn(
    const float* __restrict__ q, const float* __restrict__ k,
    const float* __restrict__ v, float* __restrict__ attR) {
  __shared__ float qs[HD];
  __shared__ float sc[LP1];
  __shared__ float red[256];
  const int b = blockIdx.x >> 3;
  const int n = blockIdx.x & 7;
  const int tid = threadIdx.x;
  const size_t base = (size_t)b * LP1 * Dd + n * HD;

  if (tid < HD) qs[tid] = q[base + (size_t)Ll * Dd + tid];
  __syncthreads();

  float lmax = -3.0e38f;
  for (int l = tid; l < LP1; l += 256) {
    const float* kr = k + base + (size_t)l * Dd;
    float s = 0.f;
#pragma unroll
    for (int j = 0; j < HD; ++j) s += qs[j] * kr[j];
    s *= SCALE;
    sc[l] = s;
    lmax = fmaxf(lmax, s);
  }
  red[tid] = lmax; __syncthreads();
  for (int off = 128; off > 0; off >>= 1) {
    if (tid < off) red[tid] = fmaxf(red[tid], red[tid + off]);
    __syncthreads();
  }
  const float m = red[0];
  __syncthreads();

  float lsum = 0.f;
  for (int l = tid; l < LP1; l += 256) {
    const float p = __expf(sc[l] - m);
    sc[l] = p;
    lsum += p;
  }
  red[tid] = lsum; __syncthreads();
  for (int off = 128; off > 0; off >>= 1) {
    if (tid < off) red[tid] += red[tid + off];
    __syncthreads();
  }
  const float inv = 1.f / red[0];
  __syncthreads();

  const int d = tid & (HD - 1);
  const int part = tid >> 6;
  float acc = 0.f;
  for (int l = part; l < LP1; l += 4)
    acc += sc[l] * v[base + (size_t)l * Dd + d];
  red[tid] = acc; __syncthreads();
  if (tid < HD)
    attR[(size_t)b * Dd + n * HD + d] =
        inv * (red[d] + red[64 + d] + red[128 + d] + red[192 + d]);
}

// ---------------------------------------------------------------------------
// Kernel 4: nodes GEMM  att(bf16,(B,L,D)) @ WO_ring + bias -> (B, D, L).
// Async double-buffered input tiles; bias fused in LDS staging; transposed
// output streamed with GLOBAL_STORE_ASYNC_FROM_LDS_B128 (coalesced along l).
// ---------------------------------------------------------------------------
__global__ __launch_bounds__(256) void nodes_gemm(
    const __bf16* __restrict__ attb, const __bf16* __restrict__ woT,
    const float* __restrict__ bias, float* __restrict__ out) {
  __shared__ __bf16 sA[2][TM * LDT];
  __shared__ __bf16 sB[2][TN * LDT];
  __shared__ float  sC[TN * 132];           // [e][l], pad 132 -> conflict-free

  const int tid  = threadIdx.x;
  const int wid  = tid >> 5;
  const int lane = tid & 31;
  int blk = blockIdx.x;
  const int eblk = blk & 7;             blk >>= 3;
  const int lblk = blk & (NLBLK_OUT - 1);
  const int b    = blk >> 5;
  const int l0   = lblk * TM;
  const int e0   = eblk * TN;

  v8f acc[4];
#pragma unroll
  for (int nt = 0; nt < 4; ++nt) acc[nt] = vzero8();

  auto issue_tiles = [&](int buf, int k0) {
#pragma unroll
    for (int i = 0; i < 2; ++i) {
      const int idx = tid + (i << 8);
      const int row = idx >> 2, ch = idx & 3;
      async_load_b128(&sA[buf][row * LDT + (ch << 3)],
                      attb + ((size_t)b * Ll + l0 + row) * Dd + k0 + (ch << 3));
    }
    const int row = tid >> 2, ch = tid & 3;
    async_load_b128(&sB[buf][row * LDT + (ch << 3)],
                    woT + (size_t)(e0 + row) * Dd + k0 + (ch << 3));
  };

  issue_tiles(0, 0);
  for (int it = 0; it < NKIT; ++it) {
    wait_async0();
    __syncthreads();
    if (it + 1 < NKIT) issue_tiles((it + 1) & 1, (it + 1) * TK);
    const int buf = it & 1;
    const v16bf af = load_afrag(&sA[buf][(wid << 4) * LDT], lane);
#pragma unroll
    for (int nt = 0; nt < 4; ++nt) {
      const v16bf bw = load_bfrag(&sB[buf][(nt << 4) * LDT], lane);
      acc[nt] = wmma_bf16(af, bw, acc[nt]);
    }
  }

  // stage C -> LDS transposed [e][l], fuse bias
  const int m0  = wid << 4;
  const int nlo = lane & 15;
  const int mhi = (lane >> 4) << 3;
#pragma unroll
  for (int nt = 0; nt < 4; ++nt) {
    const int eoff = (nt << 4) + nlo;
    const float be = bias[e0 + eoff];
#pragma unroll
    for (int r = 0; r < 8; ++r)
      sC[eoff * 132 + m0 + r + mhi] = acc[nt][r] + be;
  }
  __syncthreads();

  // async LDS -> global: 64 e-rows x 512B, coalesced along l
#pragma unroll
  for (int i = 0; i < 8; ++i) {
    const int idx  = tid + (i << 8);          // 0..2047 chunks of 16B
    const int eoff = idx >> 5;
    const int ch   = idx & 31;
    async_store_b128(out + (size_t)b * Dd * Ll + (size_t)(e0 + eoff) * Ll + l0 + (ch << 2),
                     &sC[eoff * 132 + (ch << 2)]);
  }
  // S_ENDPGM performs implicit wait-idle, draining ASYNCcnt.
}

// ---------------------------------------------------------------------------
// Kernel 5: relay output projection (8x512 @ 512x512, tiny) -> d_out tail.
// ---------------------------------------------------------------------------
__global__ __launch_bounds__(256) void relay_out(
    const float* __restrict__ attR, const float* __restrict__ WO,
    const float* __restrict__ bias, float* __restrict__ out) {
  const int gid = blockIdx.x * 256 + threadIdx.x;   // B*D = 4096
  const int e = gid & (Dd - 1);
  const int b = gid >> 9;
  const float* a = attR + (size_t)b * Dd;
  float s = bias[e];
  for (int kk = 0; kk < Dd; ++kk) s += a[kk] * WO[(size_t)kk * Dd + e];
  out[(size_t)Bx * Dd * Ll + (size_t)b * Dd + e] = s;
}

// ---------------------------------------------------------------------------
extern "C" void kernel_launch(void* const* d_in, const int* in_sizes, int n_in,
                              void* d_out, int out_size, void* d_ws, size_t ws_size,
                              hipStream_t stream) {
  const float* x   = (const float*)d_in[0];
  const float* y   = (const float*)d_in[1];
  const float* Wq  = (const float*)d_in[2];
  const float* Wk  = (const float*)d_in[3];
  const float* Wv  = (const float*)d_in[4];
  const float* WOr = (const float*)d_in[5];
  const float* bOr = (const float*)d_in[6];
  const float* WOs = (const float*)d_in[7];
  const float* bOs = (const float*)d_in[8];
  float* out = (float*)d_out;
  char* ws = (char*)d_ws;

  size_t off = 0;
  __bf16* wqT = (__bf16*)(ws + off); off += (size_t)Dd * Dd * 2;
  __bf16* wkT = (__bf16*)(ws + off); off += (size_t)Dd * Dd * 2;
  __bf16* wvT = (__bf16*)(ws + off); off += (size_t)Dd * Dd * 2;
  __bf16* woT = (__bf16*)(ws + off); off += (size_t)Dd * Dd * 2;
  __bf16* xyb = (__bf16*)(ws + off); off += (size_t)Bx * LP1 * Dd * 2;
  __bf16* attb= (__bf16*)(ws + off); off += (size_t)Bx * Ll * Dd * 2;
  float* q    = (float*)(ws + off);  off += (size_t)Bx * LP1 * Dd * 4;
  float* k    = (float*)(ws + off);  off += (size_t)Bx * LP1 * Dd * 4;
  float* v    = (float*)(ws + off);  off += (size_t)Bx * LP1 * Dd * 4;
  float* attR = (float*)(ws + off);  off += (size_t)Bx * Dd * 4;

  prep_weights<<<(Dd * Dd + 255) / 256, 256, 0, stream>>>(Wq, Wk, Wv, WOr,
                                                          wqT, wkT, wvT, woT);
  xy_to_bf16<<<Bx * 65 * 8, 256, 0, stream>>>(x, y, xyb);
  qkv_gemm<<<Bx * NLBLK_QKV * 8, 256, 0, stream>>>(xyb, wqT, wkT, wvT, q, k, v);
  node_attn<<<(Bx * NH * Ll) / 256, 256, 0, stream>>>(q, k, v, attb);
  relay_attn<<<Bx * NH, 256, 0, stream>>>(q, k, v, attR);
  nodes_gemm<<<Bx * NLBLK_OUT * 8, 256, 0, stream>>>(attb, woT, bOr, out);
  relay_out<<<(Bx * Dd) / 256, 256, 0, stream>>>(attR, WOs, bOs, out);
}
